// TopModel_49692771615096
// MI455X (gfx1250) — compile-verified, matrix operand order checked by reading, not run on hardware
//
#include <hip/hip_runtime.h>
#include <hip/hip_bf16.h>
#include <math.h>

typedef _Float16 f16;
typedef __attribute__((ext_vector_type(16))) _Float16 v16h;
typedef __attribute__((ext_vector_type(8)))  _Float16 v8h;
typedef __attribute__((ext_vector_type(8)))  float    v8f;

#define OUTN   124
#define NPIX   (124 * 124)          // 15376
#define PADN   128
#define LDF    128                  // padded row length (halves / floats)
#define NSB    64                   // 4 weight-sets * 16 batch
#define NCH    32
#define H1     126                  // conv1 output spatial
#define PI_F   3.14159265358979323846f

// fwd kernel LDS: 4 x f16[128][128] (T matrices) + 2 x f16[124..128][128] imgs
#define FWD_LDS 196608

#if defined(__AMDGCN__) && __has_builtin(__builtin_amdgcn_tensor_load_to_lds)
#define USE_TDM 1
#else
#define USE_TDM 0
#endif

// ---------------------------------------------------------------------------
// WMMA helpers (CDNA5 wave32, V_WMMA_F32_16X16X32_F16)
// ---------------------------------------------------------------------------
__device__ __forceinline__ v8f wmma_f32_f16(v16h a, v16h b, v8f c) {
  return __builtin_amdgcn_wmma_f32_16x16x32_f16(false, a, false, b, (short)0, c,
                                                false, false);
}

__device__ __forceinline__ v16h pack16(v8h lo, v8h hi) {
  return __builtin_shufflevector(lo, hi, 0, 1, 2, 3, 4, 5, 6, 7,
                                 8, 9, 10, 11, 12, 13, 14, 15);
}

// Row-contiguous operand tile: serves (a) A-tiles of row-major matrices and
// (b) B-tiles of matrices consumed transposed (incl. symmetric F).
// Lane l -> row r0+(l&15); lane half h takes k 8h..8h+7 and 16+8h..23+8h (two 16B loads).
__device__ __forceinline__ v16h load_row16(const f16* M, int ld, int r0, int k0) {
  int lane = threadIdx.x & 31;
  const f16* p = M + (size_t)(r0 + (lane & 15)) * ld + k0 + ((lane >> 4) << 3);
  return pack16(*(const v8h*)p, *(const v8h*)(p + 16));
}

// Same, with a per-lane row bound (rows >= rmax read as zero).  K span must be
// valid in storage (padded rows).
__device__ __forceinline__ v16h load_row16_rb(const f16* M, int ld, int r0, int k0,
                                              int rmax) {
  int lane = threadIdx.x & 31;
  int r = r0 + (lane & 15);
  v8h lo = {}, hi = {};
  if (r < rmax) {
    const f16* p = M + (size_t)r * ld + k0 + ((lane >> 4) << 3);
    lo = *(const v8h*)p;
    hi = *(const v8h*)(p + 16);
  }
  return pack16(lo, hi);
}

// Row-contiguous tile from padded row-major f32 (converted to f16), row bound.
__device__ __forceinline__ v16h load_row32_rb(const float* M, int ld, int r0, int k0,
                                              int rmax) {
  int lane = threadIdx.x & 31;
  int r = r0 + (lane & 15);
  v16h out = {};
  if (r < rmax) {
    const float* p = M + (size_t)r * ld + k0 + ((lane >> 4) << 3);
#pragma unroll
    for (int e = 0; e < 8; ++e) { out[e] = (f16)p[e]; out[e + 8] = (f16)p[e + 16]; }
  }
  return out;
}

// Element-gather transposed B tile from unpadded f32 (y only; tiny kernel).
__device__ __forceinline__ v16h load_gatherT32(const float* M, int ld, int k0, int n0,
                                               int kmax, int nmax) {
  int lane = threadIdx.x & 31;
  int n  = n0 + (lane & 15);
  int kb = k0 + ((lane >> 4) << 3);
  v16h r;
#pragma unroll
  for (int e = 0; e < 16; ++e) {
    int k = kb + ((e >> 3) << 4) + (e & 7);
    r[e] = (k < kmax && n < nmax) ? (f16)M[(size_t)n * ld + k] : (f16)0.f;
  }
  return r;
}

#if USE_TDM
typedef __attribute__((ext_vector_type(4))) unsigned int v4u;
typedef __attribute__((ext_vector_type(8))) int v8i;
typedef __attribute__((ext_vector_type(4))) int v4i;

// TDM D# for a 2-D copy of one padded feature map (124 rows x 128 f16) from
// global memory into LDS at byte offset lds_byte_addr (CDNA5 ISA).
__device__ __forceinline__ void tdm_load_img(unsigned lds_byte_addr, const f16* g) {
  unsigned long long ga = (unsigned long long)(uintptr_t)g;
  v4u g0 = { 1u,                                   // count=1, user mode
             lds_byte_addr,                        // lds_addr (bytes)
             (unsigned)ga,                         // global_addr[31:0]
             (unsigned)(ga >> 32) | 0x80000000u }; // global_addr[56:32] | type=2
  v8i g1 = { 0x00010000,                           // data_size = 2 bytes
             (int)(128u << 16),                    // tensor_dim0 = 128
             (int)(124u << 16),                    // tensor_dim1 = 124
             (int)(128u << 16),                    // tile_dim0   = 128
             124,                                  // tile_dim1   = 124
             128,                                  // tensor_dim0_stride = 128
             0, 0 };
  v4i gz  = { 0, 0, 0, 0 };
  v8i gz8 = { 0, 0, 0, 0, 0, 0, 0, 0 };
  __builtin_amdgcn_tensor_load_to_lds(g0, g1, gz, gz, gz8, 0);
}
#endif

// ---------------------------------------------------------------------------
// DFT matrix: F[j][k] = exp(-2*pi*i*j*k/124), symmetric, zero-padded 128x128
// ---------------------------------------------------------------------------
__global__ void gen_dft(f16* __restrict__ Fr, f16* __restrict__ Fi) {
  int idx = blockIdx.x * 256 + threadIdx.x;
  if (idx >= PADN * PADN) return;
  int j = idx >> 7, k = idx & 127;
  float fr = 0.f, fi = 0.f;
  if (j < OUTN && k < OUTN) {
    int m = (j * k) % OUTN;                       // exact phase reduction
    float ang = -2.f * PI_F * (float)m / (float)OUTN;
    fr = cosf(ang); fi = sinf(ang);
  }
  Fr[idx] = (f16)fr; Fi[idx] = (f16)fi;
}

__global__ void cast_w2(const float* __restrict__ w2, f16* __restrict__ w2h, int n) {
  int idx = blockIdx.x * 256 + threadIdx.x;
  if (idx < n) w2h[idx] = (f16)w2[idx];
}

// ---------------------------------------------------------------------------
// conv1 (3->32, 3x3 VALID, ReLU), direct VALU, output f16 [S][B][32][126][126]
// ---------------------------------------------------------------------------
__global__ void conv1_relu(const float* __restrict__ x, const float* __restrict__ w1,
                           const float* __restrict__ b1, f16* __restrict__ h) {
  size_t idx = (size_t)blockIdx.x * 256 + threadIdx.x;
  const size_t TOT = (size_t)NSB * NCH * H1 * H1;
  if (idx >= TOT) return;
  int ox = (int)(idx % H1);  size_t r = idx / H1;
  int oy = (int)(r % H1);    r /= H1;
  int oc = (int)(r % NCH);   r /= NCH;
  int b  = (int)(r % 16);
  int s  = (int)(r / 16);
  const float* xb = x + (size_t)b * 3 * 128 * 128;
  const float* wv = w1 + (size_t)(s * NCH + oc) * 27;
  float acc = b1[s * NCH + oc];
#pragma unroll
  for (int ic = 0; ic < 3; ++ic)
#pragma unroll
    for (int ky = 0; ky < 3; ++ky)
#pragma unroll
      for (int kx = 0; kx < 3; ++kx)
        acc += xb[((size_t)ic * 128 + oy + ky) * 128 + ox + kx] * wv[ic * 9 + ky * 3 + kx];
  h[idx] = (f16)(acc > 0.f ? acc : 0.f);
}

// ---------------------------------------------------------------------------
// conv2 (32->32, 3x3 VALID) via WMMA im2col GEMM, fused LRN.
// Patch staged TRANSPOSED (N x K) in LDS so B-tiles are row-contiguous.
// feat written with padded rows (ld=128, cols 124..127 zero).
// ---------------------------------------------------------------------------
__global__ void conv2_lrn(const f16* __restrict__ h, const f16* __restrict__ w2h,
                          const float* __restrict__ b2, f16* __restrict__ feat) {
  __shared__ f16   patchT[32 * 288];    // [n][k]
  __shared__ float convv[32 * 32];      // pre-LRN ch x px
  int blk = blockIdx.x;
  int ct  = blk & 3;
  int oy  = (blk >> 2) % OUTN;
  int sb  = blk / (4 * OUTN);
  int s   = sb >> 4;
  int ox  = ct * 32;
  const f16* hb = h + (size_t)sb * NCH * H1 * H1;

  for (int q = threadIdx.x; q < 288 * 32; q += 128) {
    int k = q >> 5, n = q & 31;
    int ic = k / 9, t = k - ic * 9, ky = t / 3, kx = t - ky * 3;
    int px = ox + n; if (px > OUTN - 1) px = OUTN - 1;
    patchT[n * 288 + k] = hb[((size_t)ic * H1 + (oy + ky)) * H1 + (px + kx)];
  }
  __syncthreads();

  int w = threadIdx.x >> 5, lane = threadIdx.x & 31;
  int m0 = (w >> 1) * 16, n0 = (w & 1) * 16;
  const f16* A = w2h + (size_t)s * NCH * 288;
  v8f acc = {};
#pragma unroll
  for (int k0 = 0; k0 < 288; k0 += 32) {
    v16h a = load_row16(A, 288, m0, k0);
    v16h b = load_row16(patchT, 288, n0, k0);
    acc = wmma_f32_f16(a, b, acc);
  }
  int n  = n0 + (lane & 15);
  int mb = m0 + ((lane >> 4) << 3);
#pragma unroll
  for (int v = 0; v < 8; ++v)
    convv[(mb + v) * 32 + n] = acc[v] + b2[s * NCH + mb + v];
  __syncthreads();

  for (int q = threadIdx.x; q < 32 * 32; q += 128) {
    int ch = q >> 5, px = q & 31;
    float sum = 0.f;
#pragma unroll
    for (int d = -2; d <= 2; ++d) {
      int cc = ch + d;
      if (cc >= 0 && cc < NCH) { float t = convv[cc * 32 + px]; sum += t * t; }
    }
    float val = convv[q] * powf(1.f + 2e-5f * sum, -0.75f);   // alpha/size = 2e-5
    int gx = ox + px;
    if (gx >= OUTN) val = 0.f;                                 // zero padding
    feat[((size_t)(sb * NCH + ch) * OUTN + oy) * LDF + gx] = (f16)val;
  }
}

// ---------------------------------------------------------------------------
// Forward FFT2 (transposed frame): S = F * imgT * F per z- and x-channel,
// accumulating kzz += |Z|^2 and kxz += X*conj(Z) with f32 atomics.
// Feature maps are DMA'd into LDS by the Tensor Data Mover (wave 0 issues,
// s_wait_tensorcnt, workgroup barrier releases the other 7 waves).
// One block per (sb, channel).  LDS: 128 KB T + 2 x 32 KB images.
// ---------------------------------------------------------------------------
__global__ void fwd_fft_accum(const f16* __restrict__ featZ, const f16* __restrict__ featX,
                              const f16* __restrict__ Fr, const f16* __restrict__ Fi,
                              float* __restrict__ kzz, float* __restrict__ kxzr,
                              float* __restrict__ kxzi) {
  extern __shared__ f16 smT[];
  int c  = blockIdx.x & 31;
  int sb = blockIdx.x >> 5;
  const f16* z = featZ + (size_t)(sb * NCH + c) * OUTN * LDF;
  const f16* x = featX + (size_t)(sb * NCH + c) * OUTN * LDF;
  int w = threadIdx.x >> 5, lane = threadIdx.x & 31;    // 8 waves

#if USE_TDM
  f16* imgz = smT + 65536;           // LDS byte offset 131072
  f16* imgx = smT + 81920;           // LDS byte offset 163840
  if (threadIdx.x < 32) {
    tdm_load_img(131072u, z);
    tdm_load_img(163840u, x);
    __builtin_amdgcn_s_wait_tensorcnt(0);
  }
  __syncthreads();
#endif

  // Stage 1: T = F * imgT  (4 real products: Tzr Tzi Txr Txi, 64 tiles each)
  for (int q = w; q < 256; q += 8) {
    int p  = q >> 6, t = q & 63;
    int jt = (t >> 3) << 4, ctl = (t & 7) << 4;
    const f16* A = (p & 1) ? Fi : Fr;
#if USE_TDM
    const f16* img = (p >> 1) ? imgx : imgz;
#else
    const f16* img = (p >> 1) ? x : z;
#endif
    f16* Tp = smT + ((size_t)p << 14);          // p * 128 * 128
    v8f acc = {};
#pragma unroll
    for (int k0 = 0; k0 < PADN; k0 += 32) {
      v16h a = load_row16(A, PADN, jt, k0);
      v16h b = load_row16_rb(img, LDF, ctl, k0, OUTN);   // imgT tile, contiguous
      acc = wmma_f32_f16(a, b, acc);
    }
    int n = ctl + (lane & 15), mb = jt + ((lane >> 4) << 3);
#pragma unroll
    for (int v = 0; v < 8; ++v) Tp[(mb + v) * PADN + n] = (f16)acc[v];
  }
  __syncthreads();

  f16* Tzr = smT;
  f16* Tzi = smT + 16384;
  f16* Txr = smT + 32768;
  f16* Txi = smT + 49152;

  // Stage 2: S = T * F (F symmetric -> contiguous B-tiles)
  for (int t = w; t < 64; t += 8) {
    int jt = (t >> 3) << 4, kt = (t & 7) << 4;
    v8f zrr = {}, zri = {}, zir = {}, zii = {};
    v8f xrr = {}, xri = {}, xir = {}, xii = {};
#pragma unroll
    for (int k0 = 0; k0 < PADN; k0 += 32) {
      if (k0 < PADN - 32) __builtin_prefetch(Fr + (size_t)kt * PADN + k0 + 32, 0, 0);
      v16h azr = load_row16(Tzr, PADN, jt, k0);
      v16h azi = load_row16(Tzi, PADN, jt, k0);
      v16h axr = load_row16(Txr, PADN, jt, k0);
      v16h axi = load_row16(Txi, PADN, jt, k0);
      v16h br  = load_row16(Fr, PADN, kt, k0);
      v16h bi  = load_row16(Fi, PADN, kt, k0);
      zrr = wmma_f32_f16(azr, br, zrr);  zri = wmma_f32_f16(azr, bi, zri);
      zir = wmma_f32_f16(azi, br, zir);  zii = wmma_f32_f16(azi, bi, zii);
      xrr = wmma_f32_f16(axr, br, xrr);  xri = wmma_f32_f16(axr, bi, xri);
      xir = wmma_f32_f16(axi, br, xir);  xii = wmma_f32_f16(axi, bi, xii);
    }
    int col = kt + (lane & 15), rowb = jt + ((lane >> 4) << 3);
    if (col < OUTN) {
#pragma unroll
      for (int v = 0; v < 8; ++v) {
        int row = rowb + v;
        if (row < OUTN) {
          float Zr = zrr[v] - zii[v], Zi = zri[v] + zir[v];
          float Xr = xrr[v] - xii[v], Xi = xri[v] + xir[v];
          size_t o = (size_t)sb * NPIX + (size_t)row * OUTN + col;
          atomicAdd(&kzz[o],  Zr * Zr + Zi * Zi);
          atomicAdd(&kxzr[o], Xr * Zr + Xi * Zi);
          atomicAdd(&kxzi[o], Xi * Zr - Xr * Zi);
        }
      }
    }
  }
}

// ---------------------------------------------------------------------------
// FFT2 of the (symmetric) Gaussian label y -> full complex grid Yf0.
// ---------------------------------------------------------------------------
__global__ void fft2_y(const float* __restrict__ img, const f16* __restrict__ Fr,
                       const f16* __restrict__ Fi, float* __restrict__ Yr,
                       float* __restrict__ Yi) {
  extern __shared__ f16 smY[];
  int w = threadIdx.x >> 5, lane = threadIdx.x & 31;
  for (int q = w; q < 128; q += 8) {
    int p = q >> 6, t = q & 63;
    int jt = (t >> 3) << 4, ctl = (t & 7) << 4;
    const f16* A = p ? Fi : Fr;
    f16* T = smY + ((size_t)p << 14);
    v8f acc = {};
#pragma unroll
    for (int k0 = 0; k0 < PADN; k0 += 32) {
      v16h a = load_row16(A, PADN, jt, k0);
      v16h b = load_gatherT32(img, OUTN, k0, ctl, OUTN, OUTN);
      acc = wmma_f32_f16(a, b, acc);
    }
    int n = ctl + (lane & 15), mb = jt + ((lane >> 4) << 3);
#pragma unroll
    for (int v = 0; v < 8; ++v) T[(mb + v) * PADN + n] = (f16)acc[v];
  }
  __syncthreads();
  f16* Tr = smY;
  f16* Ti = smY + 16384;
  for (int t = w; t < 64; t += 8) {
    int jt = (t >> 3) << 4, kt = (t & 7) << 4;
    v8f rr = {}, ri = {}, ir = {}, ii = {};
#pragma unroll
    for (int k0 = 0; k0 < PADN; k0 += 32) {
      v16h ar = load_row16(Tr, PADN, jt, k0);
      v16h ai = load_row16(Ti, PADN, jt, k0);
      v16h br = load_row16(Fr, PADN, kt, k0);
      v16h bi = load_row16(Fi, PADN, kt, k0);
      rr = wmma_f32_f16(ar, br, rr);  ri = wmma_f32_f16(ar, bi, ri);
      ir = wmma_f32_f16(ai, br, ir);  ii = wmma_f32_f16(ai, bi, ii);
    }
    int col = kt + (lane & 15), rowb = jt + ((lane >> 4) << 3);
    if (col < OUTN) {
#pragma unroll
      for (int v = 0; v < 8; ++v) {
        int row = rowb + v;
        if (row < OUTN) {
          Yr[(size_t)row * OUTN + col] = rr[v] - ii[v];
          Yi[(size_t)row * OUTN + col] = ri[v] + ir[v];
        }
      }
    }
  }
}

// ---------------------------------------------------------------------------
// Elementwise filter in the transposed frame, written with padded rows:
// R_T[u][v] = kxz_T(u,v) * lab_T(u,v) / (kzz_T(u,v) + lambda); pad cols = 0.
// ---------------------------------------------------------------------------
__global__ void make_R(const float* __restrict__ kzz, const float* __restrict__ kxzr,
                       const float* __restrict__ kxzi, const float* __restrict__ labR,
                       const float* __restrict__ labI, int perMap,
                       float* __restrict__ Rr, float* __restrict__ Ri) {
  int idx = blockIdx.x * 256 + threadIdx.x;
  if (idx >= NSB * OUTN * LDF) return;
  int v  = idx & (LDF - 1);
  int u  = (idx >> 7) % OUTN;
  int sb = idx / (OUTN * LDF);
  float rr = 0.f, ri = 0.f;
  if (v < OUTN) {
    int f  = u * OUTN + v;
    int g  = sb * NPIX + f;
    int li = perMap ? g : f;
    float lr = labR[li], lim = labI[li];
    float a  = 1.f / (kzz[g] + 1e-4f);
    float xr = kxzr[g], xi = kxzi[g];
    rr = (xr * lr - xi * lim) * a;
    ri = (xr * lim + xi * lr) * a;
  }
  Rr[idx] = rr;
  Ri[idx] = ri;
}

// ---------------------------------------------------------------------------
// Inverse FFT2 (real part), consumes R_T transposed -> reference-frame output:
// out = Re(conj(F) * R_ref * conj(F)) / N^2.  LDS: Ur,Ui f16[128][128] = 64 KB.
// ---------------------------------------------------------------------------
__global__ void inv_fft_response(const float* __restrict__ RrAll,
                                 const float* __restrict__ RiAll,
                                 const f16* __restrict__ Fr, const f16* __restrict__ Fi,
                                 float* __restrict__ out, float scale) {
  extern __shared__ f16 smU[];
  int sb = blockIdx.x;
  const float* Rr = RrAll + (size_t)sb * OUTN * LDF;
  const float* Ri = RiAll + (size_t)sb * OUTN * LDF;
  float* o = out + (size_t)sb * NPIX;
  int w = threadIdx.x >> 5, lane = threadIdx.x & 31;

  // Stage 1: U = conj(F) * R_ref ; Ur = Fr*Rr + Fi*Ri , Ui = Fr*Ri - Fi*Rr
  for (int q = w; q < 128; q += 8) {
    int p = q >> 6, t = q & 63;
    int jt = (t >> 3) << 4, ctl = (t & 7) << 4;
    v8f accA = {}, accB = {};
#pragma unroll
    for (int k0 = 0; k0 < PADN; k0 += 32) {
      v16h ar = load_row16(Fr, PADN, jt, k0);
      v16h ai = load_row16(Fi, PADN, jt, k0);
      v16h b1 = load_row32_rb(p ? Ri : Rr, LDF, ctl, k0, OUTN);
      v16h b2 = load_row32_rb(p ? Rr : Ri, LDF, ctl, k0, OUTN);
      accA = wmma_f32_f16(ar, b1, accA);
      accB = wmma_f32_f16(ai, b2, accB);
    }
    f16* T = smU + ((size_t)p << 14);
    float sgn = p ? -1.f : 1.f;
    int n = ctl + (lane & 15), mb = jt + ((lane >> 4) << 3);
#pragma unroll
    for (int v = 0; v < 8; ++v) T[(mb + v) * PADN + n] = (f16)(accA[v] + sgn * accB[v]);
  }
  __syncthreads();

  f16* Ur = smU;
  f16* Ui = smU + 16384;

  // Stage 2: out = (Ur*Fr + Ui*Fi) * scale  (F symmetric -> contiguous B)
  for (int t = w; t < 64; t += 8) {
    int jt = (t >> 3) << 4, kt = (t & 7) << 4;
    v8f a1 = {}, a2 = {};
#pragma unroll
    for (int k0 = 0; k0 < PADN; k0 += 32) {
      v16h ar = load_row16(Ur, PADN, jt, k0);
      v16h ai = load_row16(Ui, PADN, jt, k0);
      v16h br = load_row16(Fr, PADN, kt, k0);
      v16h bi = load_row16(Fi, PADN, kt, k0);
      a1 = wmma_f32_f16(ar, br, a1);
      a2 = wmma_f32_f16(ai, bi, a2);
    }
    int col = kt + (lane & 15), rowb = jt + ((lane >> 4) << 3);
    if (col < OUTN) {
#pragma unroll
      for (int v = 0; v < 8; ++v) {
        int row = rowb + v;
        if (row < OUTN) o[(size_t)row * OUTN + col] = (a1[v] + a2[v]) * scale;
      }
    }
  }
}

// ---------------------------------------------------------------------------
// Per-map argmax -> (row, col)  (response is in the reference frame)
// ---------------------------------------------------------------------------
__global__ void argmax_rc(const float* __restrict__ resp, int* __restrict__ rc) {
  __shared__ float sv[256];
  __shared__ int   si[256];
  int sb = blockIdx.x;
  const float* r = resp + (size_t)sb * NPIX;
  float bv = -INFINITY; int bi = 0;
  for (int i = threadIdx.x; i < NPIX; i += 256) {
    float v = r[i];
    if (v > bv || (v == bv && i < bi)) { bv = v; bi = i; }
  }
  sv[threadIdx.x] = bv; si[threadIdx.x] = bi;
  __syncthreads();
  for (int s = 128; s > 0; s >>= 1) {
    if (threadIdx.x < s) {
      if (sv[threadIdx.x + s] > sv[threadIdx.x] ||
          (sv[threadIdx.x + s] == sv[threadIdx.x] && si[threadIdx.x + s] < si[threadIdx.x])) {
        sv[threadIdx.x] = sv[threadIdx.x + s];
        si[threadIdx.x] = si[threadIdx.x + s];
      }
    }
    __syncthreads();
  }
  if (threadIdx.x == 0) { rc[2 * sb] = si[0] / OUTN; rc[2 * sb + 1] = si[0] % OUTN; }
}

// ---------------------------------------------------------------------------
// Transposed-frame shifted label:
// lab_T(u,v) = Yf0(u,v) * exp(-2*pi*i*(u*c + v*r)/N)   (note swapped r/c)
// ---------------------------------------------------------------------------
__global__ void phase_label(const float* __restrict__ Yr, const float* __restrict__ Yi,
                            const int* __restrict__ rc, float* __restrict__ labR,
                            float* __restrict__ labI) {
  int idx = blockIdx.x * 256 + threadIdx.x;
  if (idx >= NSB * NPIX) return;
  int sb = idx / NPIX, f = idx % NPIX;
  int u = f / OUTN, v = f % OUTN;
  int r = rc[2 * sb], c = rc[2 * sb + 1];
  int m = (u * c + v * r) % OUTN;
  float ang = -2.f * PI_F * (float)m / (float)OUTN;
  float pr = cosf(ang), pi = sinf(ang);
  float yr = Yr[f], yi = Yi[f];
  labR[idx] = yr * pr - yi * pi;
  labI[idx] = yr * pi + yi * pr;
}

// ---------------------------------------------------------------------------
// Host orchestration
// ---------------------------------------------------------------------------
extern "C" void kernel_launch(void* const* d_in, const int* in_sizes, int n_in,
                              void* d_out, int out_size, void* d_ws, size_t ws_size,
                              hipStream_t stream) {
  const float* tmpl = (const float*)d_in[0];
  const float* srch1 = (const float*)d_in[1];
  const float* srch2 = (const float*)d_in[2];
  const float* w1 = (const float*)d_in[3];
  const float* b1 = (const float*)d_in[4];
  const float* w2 = (const float*)d_in[5];
  const float* b2 = (const float*)d_in[6];
  const float* y  = (const float*)d_in[7];
  float* outp = (float*)d_out;

  char* ws = (char*)d_ws;
  size_t off = 0;
  auto alloc = [&](size_t bytes) {
    void* p = ws + off;
    off = (off + bytes + 255) & ~(size_t)255;
    return p;
  };

  const size_t FEATB = (size_t)NSB * NCH * OUTN * LDF * 2;   // padded f16 maps
  const size_t MAPB  = (size_t)NSB * NPIX * 4;               // unpadded f32
  const size_t RPADB = (size_t)NSB * OUTN * LDF * 4;         // padded f32

  f16*   Fr   = (f16*)alloc((size_t)PADN * PADN * 2);
  f16*   Fi   = (f16*)alloc((size_t)PADN * PADN * 2);
  f16*   w2h  = (f16*)alloc((size_t)4 * 32 * 288 * 2);
  f16*   h    = (f16*)alloc((size_t)NSB * NCH * H1 * H1 * 2);
  f16*   ft   = (f16*)alloc(FEATB);
  f16*   fs1  = (f16*)alloc(FEATB);
  f16*   fs2  = (f16*)alloc(FEATB);
  float* Yr   = (float*)alloc((size_t)NPIX * 4);
  float* Yi   = (float*)alloc((size_t)NPIX * 4);
  float* kzz  = (float*)alloc(MAPB);
  float* kxzr = (float*)alloc(MAPB);
  float* kxzi = (float*)alloc(MAPB);
  float* Rr   = (float*)alloc(RPADB);
  float* Ri   = (float*)alloc(RPADB);
  float* labR = (float*)alloc(MAPB);
  float* labI = (float*)alloc(MAPB);
  float* resp = (float*)alloc(MAPB);
  int*   rc   = (int*)alloc((size_t)NSB * 2 * 4);

  (void)hipFuncSetAttribute((const void*)fwd_fft_accum,
                            hipFuncAttributeMaxDynamicSharedMemorySize, FWD_LDS);
  (void)hipFuncSetAttribute((const void*)fft2_y,
                            hipFuncAttributeMaxDynamicSharedMemorySize, 65536);
  (void)hipFuncSetAttribute((const void*)inv_fft_response,
                            hipFuncAttributeMaxDynamicSharedMemorySize, 65536);

  // Constants
  gen_dft<<<(PADN * PADN + 255) / 256, 256, 0, stream>>>(Fr, Fi);
  cast_w2<<<(4 * 32 * 288 + 255) / 256, 256, 0, stream>>>(w2, w2h, 4 * 32 * 288);
  fft2_y<<<1, 256, 65536, stream>>>(y, Fr, Fi, Yr, Yi);

  // Features for the three inputs (h is reused sequentially)
  const float* xin[3] = { tmpl, srch1, srch2 };
  f16* feats[3] = { ft, fs1, fs2 };
  const int c1blocks = (int)(((size_t)NSB * NCH * H1 * H1 + 255) / 256);
  for (int i = 0; i < 3; ++i) {
    conv1_relu<<<c1blocks, 256, 0, stream>>>(xin[i], w1, b1, h);
    conv2_lrn<<<NSB * OUTN * 4, 128, 0, stream>>>(h, w2h, b2, feats[i]);
  }

  auto run_stage = [&](const f16* z, const f16* x, const float* lR, const float* lI,
                       int perMap, float* respOut) {
    (void)hipMemsetAsync(kzz, 0, MAPB, stream);
    (void)hipMemsetAsync(kxzr, 0, MAPB, stream);
    (void)hipMemsetAsync(kxzi, 0, MAPB, stream);
    fwd_fft_accum<<<NSB * NCH, 256, FWD_LDS, stream>>>(z, x, Fr, Fi, kzz, kxzr, kxzi);
    make_R<<<(NSB * OUTN * LDF + 255) / 256, 256, 0, stream>>>(kzz, kxzr, kxzi,
                                                               lR, lI, perMap, Rr, Ri);
    inv_fft_response<<<NSB, 256, 65536, stream>>>(Rr, Ri, Fr, Fi, respOut,
                                                  1.f / (float)NPIX);
  };

  // Stage 1: dcf(ft, fs1, yf0)   (Yf0 symmetric -> same in transposed frame)
  run_stage(ft, fs1, Yr, Yi, 0, resp);
  argmax_rc<<<NSB, 256, 0, stream>>>(resp, rc);
  phase_label<<<(NSB * NPIX + 255) / 256, 256, 0, stream>>>(Yr, Yi, rc, labR, labI);

  // Stage 2: dcf(fs1, fs2, fyf)
  run_stage(fs1, fs2, labR, labI, 1, resp);
  argmax_rc<<<NSB, 256, 0, stream>>>(resp, rc);
  phase_label<<<(NSB * NPIX + 255) / 256, 256, 0, stream>>>(Yr, Yi, rc, labR, labI);

  // Stage 3: dcf(fs2, ft, fyf2) -> final output
  run_stage(fs2, ft, labR, labI, 1, outp);
}
